// CycleFC1w1a_78091095375878
// MI455X (gfx1250) — 8-bit matrix operand order repaired and checked by reading, compiled, not run on hardware
//
#include <hip/hip_runtime.h>
#include <stdint.h>

typedef int v8i __attribute__((ext_vector_type(8)));

#define CDIM  384
#define HW    3136
#define WIDTH 56
#define KWIN  7
#define KTILES (CDIM / 64)
#define OTILES (CDIM / 16)

__global__ void pack_a_kernel(const float* __restrict__ x,
                              int8_t* __restrict__ A8, int n_total) {
    int idx = blockIdx.x * blockDim.x + threadIdx.x;
    int total = n_total * (CDIM / 4);
    if (idx >= total) return;
    int n  = idx / (CDIM / 4);
    int c4 = (idx - n * (CDIM / 4)) * 4;
    int b = n / HW;
    int s = n - b * HW;
    int w = s % WIDTH;
    uint32_t packed = 0;
#pragma unroll
    for (int i = 0; i < 4; ++i) {
        int c  = c4 + i;
        int dx = (c + KWIN / 2) % KWIN - KWIN / 2;
        int wi = w + dx;
        int8_t sg = 0;
        if (wi >= 0 && wi < WIDTH) {
            float v = x[((size_t)b * CDIM + c) * HW + (size_t)(s + dx)];
            sg = (v > 0.f) ? 1 : ((v < 0.f) ? -1 : 0);
        }
        packed |= ((uint32_t)(uint8_t)sg) << (8 * i);
    }
    *(uint32_t*)(A8 + (size_t)n * CDIM + c4) = packed;
}

__global__ void pack_b_kernel(const float* __restrict__ weight,
                              int8_t* __restrict__ Bpack) {
    int idx = blockIdx.x * blockDim.x + threadIdx.x;
    int tile = idx >> 8;
    int r    = idx & 255;
    int lane = r >> 3;
    int v    = r & 7;
    int kt = tile / OTILES;
    int ot = tile - kt * OTILES;
    int khalf = lane >> 4;
    int o  = ot * 16 + (lane & 15);
    int kb = kt * 64 + (v >> 1) * 16 + khalf * 8 + (v & 1) * 4;
    uint32_t packed = 0;
#pragma unroll
    for (int j = 0; j < 4; ++j) {
        float wv = weight[(size_t)o * CDIM + kb + j];
        int8_t sg = (wv > 0.f) ? 1 : ((wv < 0.f) ? -1 : 0);
        packed |= ((uint32_t)(uint8_t)sg) << (8 * j);
    }
    ((uint32_t*)Bpack)[idx] = packed;
}

__global__ __launch_bounds__(128)
void gemm_iu8_kernel(const int8_t* __restrict__ A8,
                     const int8_t* __restrict__ Bpack,
                     const float* __restrict__ bias,
                     float* __restrict__ out) {
    const int lane  = threadIdx.x & 31;
    const int wave  = threadIdx.x >> 5;
    const int nb    = blockIdx.x * 16;
    const int otile = blockIdx.y * 4 + wave;
    const int half  = lane >> 4;
    const int l15   = lane & 15;

    const int8_t* abase = A8 + (size_t)(nb + l15) * CDIM + (half << 3);
    const int8_t* bbase = Bpack + ((size_t)otile << 10) + ((size_t)lane << 5);

    v8i acc = {0, 0, 0, 0, 0, 0, 0, 0};
#pragma unroll
    for (int kt = 0; kt < KTILES; ++kt) {
        const int8_t* ar = abase + kt * 64;
        int2 a0 = *(const int2*)(ar);
        int2 a1 = *(const int2*)(ar + 16);
        int2 a2 = *(const int2*)(ar + 32);
        int2 a3 = *(const int2*)(ar + 48);
        v8i a;
        a[0] = a0.x; a[1] = a0.y; a[2] = a1.x; a[3] = a1.y;
        a[4] = a2.x; a[5] = a2.y; a[6] = a3.x; a[7] = a3.y;

        const int4* bp = (const int4*)(bbase + (size_t)kt * OTILES * 1024);
        int4 b0 = bp[0];
        int4 b1 = bp[1];
        v8i bf;
        bf[0] = b0.x; bf[1] = b0.y; bf[2] = b0.z; bf[3] = b0.w;
        bf[4] = b1.x; bf[5] = b1.y; bf[6] = b1.z; bf[7] = b1.w;

        acc = __builtin_amdgcn_wmma_i32_16x16x64_iu8(true, a, true, bf, acc,
                                                     false, false);
    }

    const int o  = otile * 16 + l15;
    const int b  = nb / HW;
    const int s0 = nb - b * HW;
    const float bv = bias[o];
    float* po = out + ((size_t)b * CDIM + o) * HW + s0 + half * 8;
    float4 r0, r1;
    r0.x = (float)acc[0] + bv; r0.y = (float)acc[1] + bv;
    r0.z = (float)acc[2] + bv; r0.w = (float)acc[3] + bv;
    r1.x = (float)acc[4] + bv; r1.y = (float)acc[5] + bv;
    r1.z = (float)acc[6] + bv; r1.w = (float)acc[7] + bv;
    *(float4*)(po)     = r0;
    *(float4*)(po + 4) = r1;
}

extern "C" void kernel_launch(void* const* d_in, const int* in_sizes, int n_in,
                              void* d_out, int out_size, void* d_ws, size_t ws_size,
                              hipStream_t stream) {
    const float* x      = (const float*)d_in[0];
    const float* weight = (const float*)d_in[1];
    const float* bias   = (const float*)d_in[2];
    float* out = (float*)d_out;

    const int n_total = in_sizes[0] / CDIM;

    int8_t* A8 = (int8_t*)d_ws;
    size_t a8_bytes = (size_t)n_total * CDIM;
    int8_t* Bpack = A8 + ((a8_bytes + 255) & ~(size_t)255);

    {
        int threads = n_total * (CDIM / 4);
        pack_a_kernel<<<(threads + 255) / 256, 256, 0, stream>>>(x, A8, n_total);
    }
    {
        int threads = KTILES * OTILES * 256;
        pack_b_kernel<<<threads / 256, 256, 0, stream>>>(weight, Bpack);
    }
    {
        dim3 grid(n_total / 16, OTILES / 4);
        gemm_iu8_kernel<<<grid, 128, 0, stream>>>(A8, Bpack, bias, out);
    }
}
